// GlobalSSM_8512625180868
// MI455X (gfx1250) — matrix operand order repaired and checked by reading, not compiled, not run on hardware
//
#include <hip/hip_runtime.h>
#include <hip/hip_bf16.h>
#include <math.h>

// ---------------------------------------------------------------------------
// GlobalSSM for MI455X (gfx1250):
//   bf16 WMMA GEMMs, A and B tiles staged via TDM (tensor_load_to_lds),
//   double buffered, all-scalar control flow, batched fragment loads.
//   + 3-phase chunked parallel selective scan.
// ---------------------------------------------------------------------------

typedef __bf16 bf16;
typedef __attribute__((ext_vector_type(16))) __bf16 v16bf;
typedef __attribute__((ext_vector_type(8)))  __bf16 v8bf;
typedef __attribute__((ext_vector_type(8)))  float  v8f;
typedef __attribute__((ext_vector_type(4)))  unsigned int u32x4;
typedef __attribute__((ext_vector_type(8)))  unsigned int u32x8;

#define BATCH  2
#define SEQ    4096
#define DMODEL 1024
#define NSTATE 16
#define CH     64          // scan chunks
#define CLEN   64          // steps per chunk (CH*CLEN == SEQ)

// ---------------------------------------------------------------------------
// Conversion kernels
// ---------------------------------------------------------------------------
__global__ void k_f32_to_bf16(const float* __restrict__ in, bf16* __restrict__ out, int n) {
    int i = blockIdx.x * 256 + threadIdx.x;
    if (i < n) out[i] = (bf16)in[i];
}

// W is (K x N) row-major; Wt is (N x K) row-major
__global__ void k_transpose_bf16(const float* __restrict__ W, bf16* __restrict__ Wt,
                                 int K, int N) {
    int i = blockIdx.x * 256 + threadIdx.x;
    if (i < K * N) {
        int k = i / N, n = i % N;
        Wt[(size_t)n * K + k] = (bf16)W[i];
    }
}

// ---------------------------------------------------------------------------
// TDM: DMA a (rows x 32) bf16 tile (row stride K elements) into LDS.
// D# group0 (4 SGPRs) + group1 (8 SGPRs); 2D tile => VADDR2/3 omitted (NULL).
// LDS padding: after every 16 dwords (one 64B row) insert 4 dwords -> 80B rows.
// ---------------------------------------------------------------------------
__device__ __attribute__((always_inline)) inline
void tdm_load_tile(const bf16* gaddr, void* lds, int K, int dim1, int rows) {
    unsigned long long ga = (unsigned long long)gaddr;
    u32x4 g0;
    g0[0] = 1u;                                            // count=1, user desc
    g0[1] = (unsigned)(unsigned long long)lds;             // LDS byte offset (addr[31:0])
    g0[2] = (unsigned)(ga & 0xFFFFFFFFu);                  // global_addr lo
    g0[3] = (unsigned)((ga >> 32) & 0x01FFFFFFu)           // global_addr hi (bits 56:32)
          | 0x80000000u;                                   // type=2 ("image")
    u32x8 g1;
    g1[0] = (1u << 16)        // data_size = 2 bytes
          | (1u << 20)        // pad_enable
          | (3u << 22)        // pad_interval: 16 dwords
          | (3u << 25);       // pad_amount:   4 dwords
    g1[1] = ((unsigned)K & 0xFFFFu) << 16;                 // tensor_dim0 lo16
    g1[2] = ((unsigned)K >> 16) | (((unsigned)dim1 & 0xFFFFu) << 16); // dim0 hi | dim1 lo
    g1[3] = ((unsigned)dim1 >> 16) | (32u << 16);          // dim1 hi | tile_dim0=32
    g1[4] = (unsigned)rows;                                // tile_dim1=rows, tile_dim2=0
    g1[5] = (unsigned)K;                                   // tensor_dim0_stride lo32
    g1[6] = 0u;
    g1[7] = 0u;
    asm volatile("tensor_load_to_lds %0, %1" :: "s"(g0), "s"(g1) : "memory");
}

// ---------------------------------------------------------------------------
// bf16 WMMA GEMM:  C[M,N] = A[M,K](bf16) * Bt[N,K]^T(bf16)
// Block: 4 waves (or 1 wave for small N). Wave w owns cols n0 = bx*64 + w*16.
// A tile (64x32) and B tile (rows x 32) staged in LDS by TDM, double buffered.
// All fragment loads hoisted ahead of 4 back-to-back WMMAs.
// EPI: 0 = softplus, 1 = *scale_ptr, 2 = plain
// ---------------------------------------------------------------------------
template <int EPI>
__global__ void __launch_bounds__(128) k_gemm_bf16(
        const bf16* __restrict__ A, const bf16* __restrict__ Bt,
        float* __restrict__ C, int M, int N, int K,
        const float* __restrict__ scale_ptr) {
    __shared__ __align__(16) char atile[2][64 * 80];       // padded rows: 80B stride
    __shared__ __align__(16) char btile[2][64 * 80];

    const int tid  = threadIdx.x;
    const int lane = tid & 31;
    const int wv   = __builtin_amdgcn_readfirstlane(tid) >> 5;   // scalar wave id
    const int n0   = blockIdx.x * 64 + wv * 16;                  // scalar
    const int m0   = blockIdx.y * 64;
    const bool active = (n0 < N);                                // scalar cond

    const int idx16  = lane & 15;
    const int half   = lane >> 4;
    const int kbaseA = half * 8;     // A fragment K sub-offset (elements)
    const int koffB  = half * 8;     // B fragment K sub-offset (elements)

    const bf16* ablk = A  + (size_t)m0 * K;
    const bf16* bblk = Bt + (size_t)(blockIdx.x * 64) * K;
    const int nrows  = (N - blockIdx.x * 64 < 64) ? (N - blockIdx.x * 64) : 64;
    const int nsteps = K / 32;

    v8f acc[4] = {};

    if (wv == 0) {                                         // prologue DMAs
        tdm_load_tile(ablk, &atile[0][0], K, M, 64);
        tdm_load_tile(bblk, &btile[0][0], K, N, nrows);
    }

    for (int s = 0; s < nsteps; ++s) {
        const int cur = s & 1;
        if (wv == 0) {
            if (s + 1 < nsteps) {                          // overlap next DMAs
                tdm_load_tile(ablk + (s + 1) * 32, &atile[(s + 1) & 1][0], K, M, 64);
                tdm_load_tile(bblk + (s + 1) * 32, &btile[(s + 1) & 1][0], K, N, nrows);
                __builtin_amdgcn_s_wait_tensorcnt(2);      // current pair complete
            } else {
                __builtin_amdgcn_s_wait_tensorcnt(0);
            }
        }
        __syncthreads();                                   // cur tiles visible

        if (active) {
            // ---- hoisted fragment loads: 10x ds_load_b128, single wait ----
            v16bf bfrag = {};
            v16bf afrag[4];
            {
                const char* brow = &btile[cur][(wv * 16 + idx16) * 80] + koffB * 2;
                v8bf lo = *(const v8bf*)(brow);
                v8bf hi = *(const v8bf*)(brow + 32);
#pragma unroll
                for (int j = 0; j < 8; ++j) { bfrag[j] = lo[j]; bfrag[8 + j] = hi[j]; }
            }
#pragma unroll
            for (int mt = 0; mt < 4; ++mt) {
                const char* arow = &atile[cur][(mt * 16 + idx16) * 80] + kbaseA * 2;
                v8bf lo = *(const v8bf*)(arow);            // K = k0+kbase .. +7
                v8bf hi = *(const v8bf*)(arow + 32);       // K = k0+16+kbase .. +7
                v16bf af = {};
#pragma unroll
                for (int j = 0; j < 8; ++j) { af[j] = lo[j]; af[8 + j] = hi[j]; }
                afrag[mt] = af;
            }
            // ---- 4 back-to-back WMMAs (independent accumulators) ----
#pragma unroll
            for (int mt = 0; mt < 4; ++mt)
                acc[mt] = __builtin_amdgcn_wmma_f32_16x16x32_bf16(
                    false, afrag[mt], false, bfrag, (short)0, acc[mt], false, false);
        }
        __syncthreads();                                   // reads done before reuse
    }

    if (!active) return;

    float scale = 1.0f;
    if (EPI == 1) scale = scale_ptr[0];

#pragma unroll
    for (int mt = 0; mt < 4; ++mt) {
#pragma unroll
        for (int r = 0; r < 8; ++r) {
            int m = m0 + mt * 16 + r + half * 8;
            int n = n0 + idx16;
            float v = acc[mt][r];
            if (EPI == 0) v = fmaxf(v, 0.0f) + log1pf(__expf(-fabsf(v)));  // softplus
            else if (EPI == 1) v *= scale;
            C[(size_t)m * N + n] = v;
        }
    }
}

// ---------------------------------------------------------------------------
// Chunked scan, phase 1: per-chunk local states (h_in = 0) + sum of deltas.
// ---------------------------------------------------------------------------
__global__ void k_scan_local(const float* __restrict__ delta, const float* __restrict__ x,
                             const float* __restrict__ A_log, const float* __restrict__ Bsel,
                             float* __restrict__ h_state, float* __restrict__ sumdelta) {
    const int tid = threadIdx.x;
    const int b = blockIdx.x >> 2;
    const int d = (blockIdx.x & 3) * 256 + tid;
    const int c = blockIdx.y;
    const int bd = b * DMODEL + d;
    const int t0 = c * CLEN;

    float A_row[NSTATE];
#pragma unroll
    for (int n = 0; n < NSTATE; ++n) A_row[n] = -__expf(A_log[d * NSTATE + n]);

    __shared__ float Bs[CLEN * NSTATE];
    const float* Bb = Bsel + ((size_t)b * SEQ + t0) * NSTATE;
    for (int i = tid; i < CLEN * NSTATE; i += 256) Bs[i] = Bb[i];
    __syncthreads();

    const float* dp = delta + ((size_t)b * SEQ + t0) * DMODEL + d;
    const float* xp = x     + ((size_t)b * SEQ + t0) * DMODEL + d;

    float h[NSTATE] = {};
    float sd = 0.0f;
    for (int tt = 0; tt < CLEN; ++tt) {
        float du  = dp[(size_t)tt * DMODEL];
        float xv  = xp[(size_t)tt * DMODEL];
        sd += du;
        float dux = du * xv;
#pragma unroll
        for (int n = 0; n < NSTATE; ++n)
            h[n] = h[n] * __expf(du * A_row[n]) + dux * Bs[tt * NSTATE + n];
    }

    float* hs = h_state + ((size_t)c * BATCH * DMODEL + bd) * NSTATE;
#pragma unroll
    for (int n = 0; n < NSTATE; ++n) hs[n] = h[n];
    sumdelta[(size_t)c * BATCH * DMODEL + bd] = sd;
}

// ---------------------------------------------------------------------------
// Phase 2: sequential combine over chunks. h_state[c] becomes carry-in of c.
// ---------------------------------------------------------------------------
__global__ void k_scan_combine(const float* __restrict__ A_log, float* __restrict__ h_state,
                               const float* __restrict__ sumdelta) {
    const int bd = blockIdx.x * 256 + threadIdx.x;   // 0..2047
    const int d  = bd & (DMODEL - 1);

    float A_row[NSTATE];
#pragma unroll
    for (int n = 0; n < NSTATE; ++n) A_row[n] = -__expf(A_log[d * NSTATE + n]);

    float carry[NSTATE] = {};
    for (int c = 0; c < CH; ++c) {
        float* hs = h_state + ((size_t)c * BATCH * DMODEL + bd) * NSTATE;
        float sd = sumdelta[(size_t)c * BATCH * DMODEL + bd];
#pragma unroll
        for (int n = 0; n < NSTATE; ++n) {
            float hl = hs[n];
            hs[n] = carry[n];                                  // carry entering chunk c
            carry[n] = carry[n] * __expf(A_row[n] * sd) + hl;  // propagate
        }
    }
}

// ---------------------------------------------------------------------------
// Phase 3: replay each chunk with carry-in, write y in place over delta.
// ---------------------------------------------------------------------------
__global__ void k_scan_apply(const float* __restrict__ x, const float* __restrict__ A_log,
                             float* __restrict__ delta_y, const float* __restrict__ Bsel,
                             const float* __restrict__ Csel, const float* __restrict__ h_state) {
    const int tid = threadIdx.x;
    const int b = blockIdx.x >> 2;
    const int d = (blockIdx.x & 3) * 256 + tid;
    const int c = blockIdx.y;
    const int bd = b * DMODEL + d;
    const int t0 = c * CLEN;

    float A_row[NSTATE];
#pragma unroll
    for (int n = 0; n < NSTATE; ++n) A_row[n] = -__expf(A_log[d * NSTATE + n]);

    __shared__ float Bs[CLEN * NSTATE];
    __shared__ float Cs[CLEN * NSTATE];
    const float* Bb = Bsel + ((size_t)b * SEQ + t0) * NSTATE;
    const float* Cb = Csel + ((size_t)b * SEQ + t0) * NSTATE;
    for (int i = tid; i < CLEN * NSTATE; i += 256) { Bs[i] = Bb[i]; Cs[i] = Cb[i]; }
    __syncthreads();

    float h[NSTATE];
    const float* hs = h_state + ((size_t)c * BATCH * DMODEL + bd) * NSTATE;
#pragma unroll
    for (int n = 0; n < NSTATE; ++n) h[n] = hs[n];

    const float* xp  = x       + ((size_t)b * SEQ + t0) * DMODEL + d;
    float*       dyp = delta_y + ((size_t)b * SEQ + t0) * DMODEL + d;

    for (int tt = 0; tt < CLEN; ++tt) {
        float du  = dyp[(size_t)tt * DMODEL];
        float xv  = xp[(size_t)tt * DMODEL];
        float dux = du * xv;
        float y = 0.0f;
#pragma unroll
        for (int n = 0; n < NSTATE; ++n) {
            h[n] = h[n] * __expf(du * A_row[n]) + dux * Bs[tt * NSTATE + n];
            y += h[n] * Cs[tt * NSTATE + n];
        }
        dyp[(size_t)tt * DMODEL] = y;
    }
}

// ---------------------------------------------------------------------------
// LayerNorm over D=1024; emits bf16 + h_final (t = S-1)
// ---------------------------------------------------------------------------
__global__ void k_layernorm(const float* __restrict__ yin, const float* __restrict__ gamma,
                            const float* __restrict__ beta, bf16* __restrict__ ln_bf,
                            float* __restrict__ h_final) {
    const int r = blockIdx.x;
    const int tid = threadIdx.x;
    __shared__ float red[256];

    const float* row = yin + (size_t)r * DMODEL;
    float v[4];
    float s = 0.0f;
#pragma unroll
    for (int i = 0; i < 4; ++i) { v[i] = row[tid + i * 256]; s += v[i]; }
    red[tid] = s; __syncthreads();
    for (int o = 128; o > 0; o >>= 1) { if (tid < o) red[tid] += red[tid + o]; __syncthreads(); }
    float mu = red[0] * (1.0f / DMODEL);
    __syncthreads();

    float q = 0.0f;
#pragma unroll
    for (int i = 0; i < 4; ++i) { float dd = v[i] - mu; q += dd * dd; }
    red[tid] = q; __syncthreads();
    for (int o = 128; o > 0; o >>= 1) { if (tid < o) red[tid] += red[tid + o]; __syncthreads(); }
    float rstd = rsqrtf(red[0] * (1.0f / DMODEL) + 1e-5f);

    const bool last = ((r & (SEQ - 1)) == (SEQ - 1));
    const int b = r / SEQ;
#pragma unroll
    for (int i = 0; i < 4; ++i) {
        int dd = tid + i * 256;
        float o = (v[i] - mu) * rstd * gamma[dd] + beta[dd];
        ln_bf[(size_t)r * DMODEL + dd] = (bf16)o;
        if (last) h_final[b * DMODEL + dd] = o;
    }
}

// ---------------------------------------------------------------------------
// h_proj[b][j] = sum_k h_final[b][k] * W_imp[k][j]
// ---------------------------------------------------------------------------
__global__ void k_hproj(const float* __restrict__ hf, const float* __restrict__ Wimp,
                        float* __restrict__ hp) {
    int j = blockIdx.x * 256 + threadIdx.x;
    int b = blockIdx.y;
    const float* h = hf + b * DMODEL;
    float acc = 0.0f;
    for (int k = 0; k < DMODEL; ++k) acc += h[k] * Wimp[(size_t)k * DMODEL + j];
    hp[b * DMODEL + j] = acc;
}

// ---------------------------------------------------------------------------
// raw[b][s] = dot(x[b][s][:], h_proj[b][:])
// ---------------------------------------------------------------------------
__global__ void k_raw(const float* __restrict__ x, const float* __restrict__ hp,
                      float* __restrict__ raw) {
    int r = blockIdx.x;
    int b = r / SEQ;
    int tid = threadIdx.x;
    __shared__ float red[256];
    const float* row = x + (size_t)r * DMODEL;
    const float* h = hp + b * DMODEL;
    float s = 0.0f;
#pragma unroll
    for (int i = 0; i < 4; ++i) { int dd = tid + i * 256; s += row[dd] * h[dd]; }
    red[tid] = s; __syncthreads();
    for (int o = 128; o > 0; o >>= 1) { if (tid < o) red[tid] += red[tid + o]; __syncthreads(); }
    if (tid == 0) raw[r] = red[0];
}

// ---------------------------------------------------------------------------
// Softmax over S per batch (logits = raw * 2)
// ---------------------------------------------------------------------------
__global__ void k_softmax(const float* __restrict__ raw, float* __restrict__ imp) {
    int b = blockIdx.x;
    int tid = threadIdx.x;
    __shared__ float red[256];
    const float* row = raw + (size_t)b * SEQ;

    float m = -1e30f;
    for (int i = tid; i < SEQ; i += 256) m = fmaxf(m, row[i] * 2.0f);
    red[tid] = m; __syncthreads();
    for (int o = 128; o > 0; o >>= 1) { if (tid < o) red[tid] = fmaxf(red[tid], red[tid + o]); __syncthreads(); }
    m = red[0]; __syncthreads();

    float s = 0.0f;
    for (int i = tid; i < SEQ; i += 256) s += __expf(row[i] * 2.0f - m);
    red[tid] = s; __syncthreads();
    for (int o = 128; o > 0; o >>= 1) { if (tid < o) red[tid] += red[tid + o]; __syncthreads(); }
    float inv = 1.0f / red[0];

    for (int i = tid; i < SEQ; i += 256) imp[(size_t)b * SEQ + i] = __expf(row[i] * 2.0f - m) * inv;
}

// ---------------------------------------------------------------------------
// Launch
// ---------------------------------------------------------------------------
extern "C" void kernel_launch(void* const* d_in, const int* in_sizes, int n_in,
                              void* d_out, int out_size, void* d_ws, size_t ws_size,
                              hipStream_t stream) {
    const float* x       = (const float*)d_in[0];
    const float* A_log   = (const float*)d_in[1];
    const float* W_delta = (const float*)d_in[2];
    const float* W_B     = (const float*)d_in[3];
    const float* W_C     = (const float*)d_in[4];
    const float* gamma   = (const float*)d_in[5];
    const float* beta    = (const float*)d_in[6];
    const float* W_ctx   = (const float*)d_in[7];
    const float* cscale  = (const float*)d_in[8];
    const float* W_imp   = (const float*)d_in[9];

    float* out = (float*)d_out;
    float* imp_out = out;
    float* ctx_out = out + (size_t)BATCH * SEQ;
    float* raw_out = ctx_out + (size_t)BATCH * SEQ * DMODEL;

    const int M = BATCH * SEQ;
    const int K = DMODEL;

    char* ws = (char*)d_ws;
    size_t off = 0;
    auto carve = [&](size_t bytes) { char* p = ws + off; off = (off + bytes + 255) & ~(size_t)255; return p; };

    bf16*  x_bf    = (bf16*) carve((size_t)M * K * 2);
    bf16*  wd_t    = (bf16*) carve((size_t)K * DMODEL * 2);
    bf16*  wb_t    = (bf16*) carve((size_t)NSTATE * K * 2);
    bf16*  wc_t    = (bf16*) carve((size_t)NSTATE * K * 2);
    bf16*  wctx_t  = (bf16*) carve((size_t)K * DMODEL * 2);
    float* delta_y = (float*)carve((size_t)M * DMODEL * 4);
    float* bsel    = (float*)carve((size_t)M * NSTATE * 4);
    float* csel    = (float*)carve((size_t)M * NSTATE * 4);
    bf16*  ln_bf   = (bf16*) carve((size_t)M * DMODEL * 2);
    float* h_final = (float*)carve((size_t)BATCH * DMODEL * 4);
    float* h_proj  = (float*)carve((size_t)BATCH * DMODEL * 4);
    float* h_state = (float*)carve((size_t)CH * BATCH * DMODEL * NSTATE * 4);
    float* sumdel  = (float*)carve((size_t)CH * BATCH * DMODEL * 4);
    (void)ws_size; (void)in_sizes; (void)n_in; (void)out_size;

    // 1) conversions / transposes
    k_f32_to_bf16<<<(M * K + 255) / 256, 256, 0, stream>>>(x, x_bf, M * K);
    k_transpose_bf16<<<(K * DMODEL + 255) / 256, 256, 0, stream>>>(W_delta, wd_t, K, DMODEL);
    k_transpose_bf16<<<(K * NSTATE + 255) / 256, 256, 0, stream>>>(W_B, wb_t, K, NSTATE);
    k_transpose_bf16<<<(K * NSTATE + 255) / 256, 256, 0, stream>>>(W_C, wc_t, K, NSTATE);
    k_transpose_bf16<<<(K * DMODEL + 255) / 256, 256, 0, stream>>>(W_ctx, wctx_t, K, DMODEL);

    // 2) projections via WMMA + TDM
    k_gemm_bf16<0><<<dim3(DMODEL / 64, M / 64), dim3(128), 0, stream>>>(
        x_bf, wd_t, delta_y, M, DMODEL, K, nullptr);
    k_gemm_bf16<2><<<dim3(1, M / 64), dim3(32), 0, stream>>>(
        x_bf, wb_t, bsel, M, NSTATE, K, nullptr);
    k_gemm_bf16<2><<<dim3(1, M / 64), dim3(32), 0, stream>>>(
        x_bf, wc_t, csel, M, NSTATE, K, nullptr);

    // 3) chunked parallel selective scan
    k_scan_local<<<dim3(BATCH * (DMODEL / 256), CH), 256, 0, stream>>>(
        delta_y, x, A_log, bsel, h_state, sumdel);
    k_scan_combine<<<BATCH * DMODEL / 256, 256, 0, stream>>>(A_log, h_state, sumdel);
    k_scan_apply<<<dim3(BATCH * (DMODEL / 256), CH), 256, 0, stream>>>(
        x, A_log, delta_y, bsel, csel, h_state);

    // 4) layernorm -> bf16 + h_final
    k_layernorm<<<M, 256, 0, stream>>>(delta_y, gamma, beta, ln_bf, h_final);

    // 5) importance path
    k_hproj<<<dim3(DMODEL / 256, BATCH), 256, 0, stream>>>(h_final, W_imp, h_proj);
    k_raw<<<M, 256, 0, stream>>>(x, h_proj, raw_out);
    k_softmax<<<BATCH, 256, 0, stream>>>(raw_out, imp_out);

    // 6) context = LN(ssm) @ W_ctx * context_scale
    k_gemm_bf16<1><<<dim3(DMODEL / 64, M / 64), dim3(128), 0, stream>>>(
        ln_bf, wctx_t, ctx_out, M, DMODEL, K, cscale);
}
